// EncDec_55078660604316
// MI455X (gfx1250) — matrix operand order repaired and checked by reading, not compiled, not run on hardware
//
#include <hip/hip_runtime.h>
#include <stdint.h>

// ---------------------------------------------------------------------------
// Problem dims
// ---------------------------------------------------------------------------
#define B_  32
#define T_  64
#define S_  128
#define H_  512
#define E_  512
#define V_  32000

typedef __attribute__((ext_vector_type(16))) __bf16 v16bf;
typedef __attribute__((ext_vector_type(8)))  float  v8f;
typedef __attribute__((ext_vector_type(4)))  unsigned int v4u;
typedef __attribute__((ext_vector_type(8)))  int    v8i;
typedef __attribute__((ext_vector_type(4)))  int    v4i;

union Frag { v16bf v; unsigned int u[8]; };

__device__ __forceinline__ unsigned short f2bf(float x) {
    return (unsigned short)(__float_as_uint(x) >> 16);   // truncate
}
__device__ __forceinline__ float bf2f(unsigned short u) {
    return __uint_as_float(((unsigned int)u) << 16);
}
__device__ __forceinline__ float sigmoidf_(float x) {
    return 1.0f / (1.0f + __expf(-x));
}

// ---------------------------------------------------------------------------
// WMMA fragment loaders (CDNA5 16-bit layouts, CDNA5 ISA §7.12.2)
// A-matrix 16x32: lane L (half=L>>4, m=L&15), VGPR r holds K = kb,kb+1 with
//   kb = (r<4 ? 2r : 2r+8) + 8*half  -> always even => one 32-bit load.
// ---------------------------------------------------------------------------
__device__ __forceinline__ v16bf load_a_frag(const unsigned short* A,
                                             int m0, int k0, int ldk, int lane) {
    int half = lane >> 4;
    int m    = m0 + (lane & 15);
    const unsigned short* row = A + (size_t)m * ldk + k0;
    Frag f;
#pragma unroll
    for (int r = 0; r < 8; ++r) {
        int kb = ((r < 4) ? (2 * r) : (2 * r + 8)) + 8 * half;
        f.u[r] = *(const unsigned int*)(row + kb);
    }
    return f.v;
}

// B-matrix 32x16 where B[k][n] = W[n0+n][k0+k], W stored [N][K] row-major.
// Lane L: n=L&15, half=L>>4; VGPR r holds K = kb, kb+1, kb = (r<4 ? 2r : 2r+8) + 8*half.
__device__ __forceinline__ v16bf load_b_frag(const unsigned short* W,
                                             int n0, int k0, int ldk, int lane) {
    int half = lane >> 4;
    int n    = n0 + (lane & 15);
    const unsigned short* row = W + (size_t)n * ldk + k0;
    Frag f;
#pragma unroll
    for (int r = 0; r < 8; ++r) f.u[r] = *(const unsigned int*)(row + ((r < 4) ? (2 * r) : (2 * r + 8)) + 8 * half);
    return f.v;
}

__device__ __forceinline__ v8f wmma_bf16(v16bf a, v16bf b, v8f c) {
    return __builtin_amdgcn_wmma_f32_16x16x32_bf16(false, a, false, b,
                                                   (short)0, c, false, false);
}

// ---------------------------------------------------------------------------
// Tensor Data Mover: build D# (groups 0/1 per CDNA5 ISA)
// 2D tile of bf16 (data_size code 1 = 2 bytes). This toolchain exposes the
// 6-arg builtin: (v4u g0, v8i g1, v4i g2, v4i g3, v8i g4, int cpol).
// Groups 2/3 (and the extra group) zero-filled for a 2D tile.
// ---------------------------------------------------------------------------
__device__ __forceinline__ void tdm_load_tile_2d(unsigned int lds_off,
                                                 const unsigned short* gptr,
                                                 unsigned int tensor_d0,
                                                 unsigned int tensor_d1,
                                                 unsigned int tile_d0,
                                                 unsigned int tile_d1,
                                                 unsigned int stride0) {
    unsigned long long ga = (unsigned long long)(uintptr_t)gptr;
    v4u g0;
    g0.x = 1u;                                          // count=1, user mode
    g0.y = lds_off;                                     // lds_addr (bytes)
    g0.z = (unsigned int)(ga & 0xFFFFFFFFu);            // global_addr[31:0]
    g0.w = (unsigned int)((ga >> 32) & 0x1FFFFFFu)      // global_addr[56:32]
         | (2u << 30);                                  // type=2 ("image")
    v8i g1;
    g1[0] = (int)(1u << 16);                            // data_size=1 (2B)
    g1[1] = (int)(tensor_d0 << 16);                     // tensor_dim0 lo16 @63:48
    g1[2] = (int)((tensor_d0 >> 16) | (tensor_d1 << 16));
    g1[3] = (int)((tensor_d1 >> 16) | (tile_d0 << 16)); // tile_dim0 @127:112
    g1[4] = (int)(tile_d1 & 0xFFFFu);                   // tile_dim1; tile_dim2=0
    g1[5] = (int)stride0;                               // dim0_stride lo32
    g1[6] = 0;
    g1[7] = 0;
    v4i z4 = {0, 0, 0, 0};
    v8i z8 = {0, 0, 0, 0, 0, 0, 0, 0};
    __builtin_amdgcn_tensor_load_to_lds(g0, g1, z4, z4, z8, 0);
}

// ---------------------------------------------------------------------------
// Kernel 0: fp32 -> bf16 convert
// ---------------------------------------------------------------------------
__global__ void cvt_bf16_kernel(const float* __restrict__ src,
                                unsigned short* __restrict__ dst, int n) {
    int i = blockIdx.x * blockDim.x + threadIdx.x;
    if (i < n) dst[i] = f2bf(src[i]);
}

// ---------------------------------------------------------------------------
// Kernel 1: sourceHtrans[b][s][h] = sum_d sourceH[b][s][d] * W_att[h][d]
// One WG (256 thr, 8 waves) per batch b. M=128 (s), N=512 (h), K=1024.
// ---------------------------------------------------------------------------
__global__ void att_pre_kernel(const unsigned short* __restrict__ SHbf,
                               const unsigned short* __restrict__ Wattbf,
                               unsigned short* __restrict__ SHT) {
    int b    = blockIdx.x;
    int lane = threadIdx.x & 31;
    int wave = threadIdx.x >> 5;
    const unsigned short* A   = SHbf + (size_t)b * S_ * (2 * H_);
    unsigned short*       out = SHT  + (size_t)b * S_ * H_;

    for (int tile = wave; tile < 256; tile += 8) {      // 8 m-sub x 32 n-sub
        int ms = tile & 7, ns = tile >> 3;
        v8f acc = {0.f, 0.f, 0.f, 0.f, 0.f, 0.f, 0.f, 0.f};
#pragma unroll 4
        for (int kc = 0; kc < 32; ++kc) {
            v16bf af = load_a_frag(A, ms * 16, kc * 32, 2 * H_, lane);
            v16bf bf = load_b_frag(Wattbf, ns * 16, kc * 32, 2 * H_, lane);
            acc = wmma_bf16(af, bf, acc);
        }
        int half = lane >> 4, n = ns * 16 + (lane & 15);
#pragma unroll
        for (int r = 0; r < 8; ++r) {
            int m = ms * 16 + r + 8 * half;
            out[(size_t)m * H_ + n] = f2bf(acc[r]);
        }
    }
}

// ---------------------------------------------------------------------------
// Kernel 2: persistent decoder recurrence. 1 WG, 512 threads = 16 wave32s.
// Dynamic LDS layout (bytes):
//   X      [32][1536] bf16  @ 0        (x_t | prevF | h)         98304
//   G/X2   [32][2048] bf16  @ 98304    (gates, then [h|ctx])    131072
//   scores [32][128]  f32   @ 229376                             16384
// ---------------------------------------------------------------------------
__global__ void __launch_bounds__(512) decoder_kernel(
        const float* __restrict__ inputTarget,  // [B][T][E]
        const float* __restrict__ h0,           // [B][H]
        const float* __restrict__ c0,           // [B][H]
        const float* __restrict__ sourceH,      // [B][S][2H] fp32
        const unsigned short* __restrict__ Wihbf,   // [2048][1024]
        const unsigned short* __restrict__ Whhbf,   // [2048][512]
        const float* __restrict__ b_ih,         // [2048]
        const float* __restrict__ b_hh,         // [2048]
        const unsigned short* __restrict__ SHT, // [B][S][H] bf16
        const unsigned short* __restrict__ Wfhbf,   // [512][1536]
        const float* __restrict__ b_fh,         // [512]
        const int* __restrict__ lengthsSource,  // [B]
        unsigned short* __restrict__ FHbf)      // [B*T][E] bf16
{
    extern __shared__ char smraw[];
    unsigned short* X      = (unsigned short*)smraw;            // [32][1536]
    unsigned short* G      = X + 32 * 1536;                     // [32][2048]
    unsigned short* X2     = G;                                 // alias (reuse)
    float*          scores = (float*)(G + 32 * 2048);           // [32][128]

    const int tid  = threadIdx.x;
    const int lane = tid & 31;
    const int wave = tid >> 5;
    const int bb   = tid >> 4;          // 0..31 : batch row owned
    const int j0   = (tid & 15) * 32;   // 32 hidden slots owned

    // ---- init: cell state in registers, h0 -> X, prevF = 0 ----
    float creg[32];
#pragma unroll
    for (int q = 0; q < 32; ++q) creg[q] = c0[bb * H_ + j0 + q];
#pragma unroll
    for (int q = 0; q < 32; ++q) {
        X[bb * 1536 + 1024 + j0 + q] = f2bf(h0[bb * H_ + j0 + q]);
        X[bb * 1536 + 512  + j0 + q] = 0;                   // bf16 zero
    }
    __syncthreads();

    for (int t = 0; t < T_; ++t) {
        // ---- stage x_t (convert to bf16) ----
#pragma unroll
        for (int q = 0; q < 32; ++q)
            X[bb * 1536 + j0 + q] =
                f2bf(inputTarget[((size_t)bb * T_ + t) * E_ + j0 + q]);
        __syncthreads();

        // ---- gates = X(32x1536) x [W_ih|W_hh]^T -> [32][2048] ----
        for (int tile = wave; tile < 256; tile += 16) {     // 2 m-sub x 128 n-sub
            int ms = tile & 1, ns = tile >> 1;
            v8f acc = {0.f, 0.f, 0.f, 0.f, 0.f, 0.f, 0.f, 0.f};
            for (int kc = 0; kc < 48; ++kc) {
                int k0 = kc * 32;
                v16bf af = load_a_frag(X, ms * 16, k0, 1536, lane);
                v16bf bf = (k0 < 1024)
                    ? load_b_frag(Wihbf, ns * 16, k0,        1024, lane)
                    : load_b_frag(Whhbf, ns * 16, k0 - 1024,  512, lane);
                acc = wmma_bf16(af, bf, acc);
            }
            int half = lane >> 4, n = ns * 16 + (lane & 15);
#pragma unroll
            for (int r = 0; r < 8; ++r) {
                int m = ms * 16 + r + 8 * half;
                G[m * 2048 + n] = f2bf(acc[r]);
            }
        }
        __syncthreads();

        // ---- LSTM cell (gate order i,f,g,o). Read gates first, then free G ----
        float gi[32], gf[32], gg[32], go[32];
#pragma unroll
        for (int q = 0; q < 32; ++q) {
            int j = j0 + q;
            gi[q] = bf2f(G[bb * 2048 + j]);
            gf[q] = bf2f(G[bb * 2048 + 512 + j]);
            gg[q] = bf2f(G[bb * 2048 + 1024 + j]);
            go[q] = bf2f(G[bb * 2048 + 1536 + j]);
        }
        __syncthreads();    // everyone done reading G; X2 may now alias it
#pragma unroll
        for (int q = 0; q < 32; ++q) {
            int j = j0 + q;
            float I = sigmoidf_(gi[q] + b_ih[j] + b_hh[j]);
            float F = sigmoidf_(gf[q] + b_ih[512 + j] + b_hh[512 + j]);
            float Gv = tanhf(gg[q] + b_ih[1024 + j] + b_hh[1024 + j]);
            float O = sigmoidf_(go[q] + b_ih[1536 + j] + b_hh[1536 + j]);
            creg[q] = F * creg[q] + I * Gv;
            float h = O * tanhf(creg[q]);
            unsigned short hb = f2bf(h);
            X [bb * 1536 + 1024 + j] = hb;   // h slot for next gates GEMM
            X2[bb * 1536 + j]        = hb;   // h slot for fh GEMM
        }
        __syncthreads();

        // ---- attention scores: per-(b,s) dot of h with sourceHtrans + mask ----
#pragma unroll
        for (int e = 0; e < 8; ++e) {
            int p = tid + e * 512;           // 0..4095
            int b = p >> 7, s = p & 127;
            const unsigned short* hrow = X + b * 1536 + 1024;
            const unsigned short* srow = SHT + ((size_t)b * S_ + s) * H_;
            float acc = 0.f;
            for (int k = 0; k < H_; ++k) acc += bf2f(hrow[k]) * bf2f(srow[k]);
            float maskv = (s < lengthsSource[b]) ? 0.f : -1024.f;
            scores[b * S_ + s] = acc + maskv;
        }
        __syncthreads();

        // ---- softmax over s (thread b handles row b) ----
        if (tid < B_) {
            float mx = -1e30f;
            for (int s = 0; s < S_; ++s) mx = fmaxf(mx, scores[tid * S_ + s]);
            float sum = 0.f;
            for (int s = 0; s < S_; ++s) {
                float ev = __expf(scores[tid * S_ + s] - mx);
                scores[tid * S_ + s] = ev;
                sum += ev;
            }
            float inv = 1.f / sum;
            for (int s = 0; s < S_; ++s) scores[tid * S_ + s] *= inv;
        }
        __syncthreads();

        // ---- ctx[b][d] = sum_s attn[b][s] * sourceH[b][s][d] -> X2 ctx slot ----
#pragma unroll 2
        for (int e = 0; e < 64; ++e) {
            int p = tid + e * 512;           // 0..32767
            int b = p >> 10, d = p & 1023;
            float acc = 0.f;
            for (int s = 0; s < S_; ++s)
                acc += scores[b * S_ + s] *
                       sourceH[((size_t)b * S_ + s) * (2 * H_) + d];
            X2[b * 1536 + 512 + d] = f2bf(acc);
        }
        __syncthreads();

        // ---- fh = tanh(X2(32x1536) x W_fh^T + b_fh) -> prevF slot + FHbf ----
        for (int tile = wave; tile < 64; tile += 16) {      // 2 m-sub x 32 n-sub
            int ms = tile & 1, ns = tile >> 1;
            v8f acc = {0.f, 0.f, 0.f, 0.f, 0.f, 0.f, 0.f, 0.f};
            for (int kc = 0; kc < 48; ++kc) {
                v16bf af = load_a_frag(X2, ms * 16, kc * 32, 1536, lane);
                v16bf bf = load_b_frag(Wfhbf, ns * 16, kc * 32, 1536, lane);
                acc = wmma_bf16(af, bf, acc);
            }
            int half = lane >> 4, n = ns * 16 + (lane & 15);
#pragma unroll
            for (int r = 0; r < 8; ++r) {
                int m = ms * 16 + r + 8 * half;             // = batch index
                unsigned short fb = f2bf(tanhf(acc[r] + b_fh[n]));
                X[m * 1536 + 512 + n] = fb;                 // prevF for t+1
                FHbf[((size_t)m * T_ + t) * E_ + n] = fb;   // row b*T+t
            }
        }
        __syncthreads();
    }
}

// ---------------------------------------------------------------------------
// Kernel 3: logits[2048][32000] = FH(bf16) x W_out(bf16)^T + b_out.
// 128x128 tile per WG (256 thr, 8 waves: 2x4). K=512 in 16 chunks of 32.
// Double-buffered TDM pipeline: tensor ops from one wave complete in order,
// so after issuing chunk kc+1, TENSORcnt <= 2 guarantees chunk kc is in LDS.
// LDS: ping-pong buffers, 2 x (A 128x32 + B 128x32) bf16 = 32 KB.
// ---------------------------------------------------------------------------
__global__ void __launch_bounds__(256) out_gemm_kernel(
        const unsigned short* __restrict__ FHbf,    // [2048][512]
        const unsigned short* __restrict__ Woutbf,  // [32000][512]
        const float* __restrict__ b_out,            // [32000]
        float* __restrict__ out)                    // [2048][32000]
{
    __shared__ unsigned short tiles[4 * 128 * 32];
    // buffer b (b=0,1): A at elem b*8192 (byte b*16384), B at elem b*8192+4096.

    const int tid  = threadIdx.x;
    const int lane = tid & 31;
    const int wave = tid >> 5;
    const int wr   = wave >> 2;     // 0..1 -> 64 rows
    const int wc   = wave & 3;      // 0..3 -> 32 cols
    const int n0g  = blockIdx.x * 128;
    const int m0g  = blockIdx.y * 128;

    v8f acc[4][2];
#pragma unroll
    for (int i = 0; i < 4; ++i)
#pragma unroll
        for (int j = 0; j < 2; ++j)
            acc[i][j] = (v8f){0.f, 0.f, 0.f, 0.f, 0.f, 0.f, 0.f, 0.f};

    // Pre-issue chunk 0 into buffer 0.
    if (wave == 0) {
        tdm_load_tile_2d(0u,     FHbf   + (size_t)m0g * E_, E_, 2048u,
                         32u, 128u, E_);
        tdm_load_tile_2d(8192u,  Woutbf + (size_t)n0g * E_, E_, (unsigned)V_,
                         32u, 128u, E_);
    }

    for (int kc = 0; kc < 16; ++kc) {
        int cur = kc & 1;
        if (wave == 0) {
            if (kc + 1 < 16) {
                int nxt = (kc + 1) & 1;
                tdm_load_tile_2d((unsigned)(nxt * 16384),
                                 FHbf + (size_t)m0g * E_ + (kc + 1) * 32,
                                 E_, 2048u, 32u, 128u, E_);
                tdm_load_tile_2d((unsigned)(nxt * 16384 + 8192),
                                 Woutbf + (size_t)n0g * E_ + (kc + 1) * 32,
                                 E_, (unsigned)V_, 32u, 128u, E_);
                __builtin_amdgcn_s_wait_tensorcnt(2);   // chunk kc complete
            } else {
                __builtin_amdgcn_s_wait_tensorcnt(0);
            }
        }
        __syncthreads();

        if (lane == 0 && kc + 2 < 16) {     // hint chunk kc+2 W tile into GL2
            __builtin_prefetch(Woutbf + (size_t)n0g * E_ + (kc + 2) * 32, 0, 1);
        }

        const unsigned short* Atile = tiles + cur * 8192;
        const unsigned short* Btile = tiles + cur * 8192 + 4096;
        v16bf afr[4], bfr[2];
#pragma unroll
        for (int i = 0; i < 4; ++i)
            afr[i] = load_a_frag(Atile, wr * 64 + i * 16, 0, 32, lane);
#pragma unroll
        for (int j = 0; j < 2; ++j)
            bfr[j] = load_b_frag(Btile, wc * 32 + j * 16, 0, 32, lane);
#pragma unroll
        for (int i = 0; i < 4; ++i)
#pragma unroll
            for (int j = 0; j < 2; ++j)
                acc[i][j] = wmma_bf16(afr[i], bfr[j], acc[i][j]);
        __syncthreads();    // buffer `cur` free for chunk kc+2's TDM next iter
    }

    const int half = lane >> 4, nl = lane & 15;
#pragma unroll
    for (int i = 0; i < 4; ++i) {
#pragma unroll
        for (int j = 0; j < 2; ++j) {
            int ng = n0g + wc * 32 + j * 16 + nl;
            float bias = b_out[ng];
#pragma unroll
            for (int r = 0; r < 8; ++r) {
                int mg = m0g + wr * 64 + i * 16 + r + 8 * half;
                out[(size_t)mg * V_ + ng] = acc[i][j][r] + bias;
            }
        }
    }
}

// ---------------------------------------------------------------------------
// Host launcher
// ---------------------------------------------------------------------------
static inline void cvt(const float* s, unsigned short* d, int n, hipStream_t st) {
    cvt_bf16_kernel<<<(n + 255) / 256, 256, 0, st>>>(s, d, n);
}

extern "C" void kernel_launch(void* const* d_in, const int* in_sizes, int n_in,
                              void* d_out, int out_size, void* d_ws, size_t ws_size,
                              hipStream_t stream) {
    (void)in_sizes; (void)n_in; (void)out_size; (void)ws_size;

    const float* inputTarget   = (const float*)d_in[0];   // [B][T][E]
    const float* h0            = (const float*)d_in[1];   // [1][B][H]
    const float* c0            = (const float*)d_in[2];   // [1][B][H]
    const float* sourceH       = (const float*)d_in[3];   // [B][S][2H]
    const float* W_ih          = (const float*)d_in[4];   // [2048][1024]
    const float* W_hh          = (const float*)d_in[5];   // [2048][512]
    const float* b_ih          = (const float*)d_in[6];
    const float* b_hh          = (const float*)d_in[7];
    const float* W_att         = (const float*)d_in[8];   // [512][1024]
    const float* W_fh          = (const float*)d_in[9];   // [512][1536]
    const float* b_fh          = (const float*)d_in[10];
    const float* W_out         = (const float*)d_in[11];  // [32000][512]
    const float* b_out         = (const float*)d_in[12];
    const int*   lengthsSource = (const int*)d_in[14];

    // -------- workspace carve (bf16 buffers) --------
    char* ws = (char*)d_ws;
    size_t off = 0;
    unsigned short* SHbf   = (unsigned short*)(ws + off); off += (size_t)B_*S_*2*H_*2;   // 8 MB
    unsigned short* Wattbf = (unsigned short*)(ws + off); off += (size_t)H_*2*H_*2;      // 1 MB
    unsigned short* SHT    = (unsigned short*)(ws + off); off += (size_t)B_*S_*H_*2;     // 4 MB
    unsigned short* Wihbf  = (unsigned short*)(ws + off); off += (size_t)4*H_*(E_+H_)*2; // 4 MB
    unsigned short* Whhbf  = (unsigned short*)(ws + off); off += (size_t)4*H_*H_*2;      // 2 MB
    unsigned short* Wfhbf  = (unsigned short*)(ws + off); off += (size_t)E_*3*H_*2;      // 1.5 MB
    unsigned short* Woutbf = (unsigned short*)(ws + off); off += (size_t)V_*E_*2;        // 32.8 MB
    unsigned short* FHbf   = (unsigned short*)(ws + off); off += (size_t)B_*T_*E_*2;     // 2 MB

    // -------- stage 0: fp32 -> bf16 --------
    cvt(sourceH, SHbf,   B_*S_*2*H_,   stream);
    cvt(W_att,   Wattbf, H_*2*H_,      stream);
    cvt(W_ih,    Wihbf,  4*H_*(E_+H_), stream);
    cvt(W_hh,    Whhbf,  4*H_*H_,      stream);
    cvt(W_fh,    Wfhbf,  E_*3*H_,      stream);
    cvt(W_out,   Woutbf, V_*E_,        stream);

    // -------- stage 1: attention precompute --------
    att_pre_kernel<<<B_, 256, 0, stream>>>(SHbf, Wattbf, SHT);

    // -------- stage 2: sequential decoder (single persistent WG) --------
    size_t dec_lds = (size_t)32*1536*2 + (size_t)32*2048*2 + (size_t)32*128*4; // 245760 B
    decoder_kernel<<<1, 512, dec_lds, stream>>>(
        inputTarget, h0, c0, sourceH, Wihbf, Whhbf, b_ih, b_hh,
        SHT, Wfhbf, b_fh, lengthsSource, FHbf);

    // -------- stage 3: big output GEMM via WMMA + double-buffered TDM --------
    dim3 grid(V_ / 128, (B_ * T_) / 128);   // 250 x 16
    out_gemm_kernel<<<grid, 256, 0, stream>>>(FHbf, Woutbf, b_out, (float*)d_out);
}
